// SequenceVAE_31473520345458
// MI455X (gfx1250) — matrix operand order repaired and checked by reading, not compiled, not run on hardware
//
#include <hip/hip_runtime.h>
#include <hip/hip_bf16.h>

typedef __bf16 bf16;
typedef __attribute__((ext_vector_type(16))) __bf16 v16bf;
typedef __attribute__((ext_vector_type(8)))  float   v8f;
typedef __attribute__((ext_vector_type(4)))  int     v4i;

union BFr { uint4 u[2]; v16bf v; };

#if defined(__has_builtin)
#  if __has_builtin(__builtin_amdgcn_global_load_async_to_lds_b128)
#    define HAVE_ASYNC_COPY 1
#  endif
#  if __has_builtin(__builtin_amdgcn_s_wait_asynccnt)
#    define HAVE_WAIT_ASYNC 1
#  endif
#endif

#ifdef HAVE_ASYNC_COPY
typedef __attribute__((address_space(1))) v4i gv4i;   // global (device) int4
typedef __attribute__((address_space(3))) v4i lv4i;   // LDS int4
#endif

#define BB  512
#define TT  64
#define AA  64
#define SDD 64
#define HH  1024
#define LL  256
#define H3  3072

// ---------------- elementwise helpers ----------------

__global__ __launch_bounds__(256) void k_zero(unsigned int* p, int n) {
  int i = blockIdx.x * 256 + threadIdx.x;
  if (i < n) p[i] = 0u;
}

__global__ __launch_bounds__(256) void k_cvt(const float* __restrict__ s,
                                             bf16* __restrict__ d, int n) {
  int i = blockIdx.x * 256 + threadIdx.x;
  if (i < n) d[i] = (bf16)s[i];
}

__global__ __launch_bounds__(256) void k_copy_bf(const bf16* __restrict__ s, int sld,
                                                 bf16* __restrict__ d, int dld,
                                                 int rows, int cols) {
  int i = blockIdx.x * 256 + threadIdx.x;
  if (i >= rows * cols) return;
  int r = i / cols, c = i - r * cols;
  d[(long)r * dld + c] = s[(long)r * sld + c];
}

// GRU gate fusion: gi = x@Wih^T (no bias), gh = h@Whh^T (no bias); biases added here.
__global__ __launch_bounds__(256) void k_gru_gate(const float* __restrict__ gi,
                                                  const float* __restrict__ gh,
                                                  const float* __restrict__ bih,
                                                  const float* __restrict__ bhh,
                                                  float* __restrict__ h,
                                                  bf16* __restrict__ hbf) {
  int i = blockIdx.x * 256 + threadIdx.x;
  if (i >= BB * HH) return;
  int b = i / HH, j = i - b * HH;
  long o = (long)b * H3;
  float ir = gi[o + j]            + bih[j];
  float iz = gi[o + HH + j]       + bih[HH + j];
  float in = gi[o + 2 * HH + j]   + bih[2 * HH + j];
  float hr = gh[o + j]            + bhh[j];
  float hz = gh[o + HH + j]       + bhh[HH + j];
  float hn = gh[o + 2 * HH + j]   + bhh[2 * HH + j];
  float r = 1.f / (1.f + __expf(-(ir + hr)));
  float z = 1.f / (1.f + __expf(-(iz + hz)));
  float n = tanhf(in + r * hn);
  float hv = (1.f - z) * n + z * h[(long)b * HH + j];
  h[(long)b * HH + j]   = hv;
  hbf[(long)b * HH + j] = (bf16)hv;
}

// z = mu + eps*exp(0.5*logvar); build bf16 concat [z, state]  (B x (L+SD))
__global__ __launch_bounds__(256) void k_reparam(const float* __restrict__ mu,
                                                 const float* __restrict__ lv,
                                                 const float* __restrict__ eps,
                                                 const float* __restrict__ state,
                                                 bf16* __restrict__ zcat) {
  const int cols = LL + SDD;
  int i = blockIdx.x * 256 + threadIdx.x;
  if (i >= BB * cols) return;
  int b = i / cols, c = i - b * cols;
  float v;
  if (c < LL) {
    float m = mu[(long)b * LL + c];
    float l = lv[(long)b * LL + c];
    v = m + eps[(long)b * LL + c] * __expf(0.5f * l);
  } else {
    v = state[(long)b * SDD + (c - LL)];
  }
  zcat[i] = (bf16)v;
}

__global__ __launch_bounds__(256) void k_initinp(const float* __restrict__ initA,
                                                 bf16* __restrict__ inp) {
  int i = blockIdx.x * 256 + threadIdx.x;
  if (i >= BB * AA) return;
  inp[i] = (bf16)initA[i % AA];
}

// teacher forcing: next input = tf_mask[t] ? target : generated action
__global__ __launch_bounds__(256) void k_nextinp(const float* __restrict__ rec,
                                                 const float* __restrict__ aseq,
                                                 const int* __restrict__ tf, int t,
                                                 bf16* __restrict__ inp) {
  int i = blockIdx.x * 256 + threadIdx.x;
  if (i >= BB * AA) return;
  int b = i / AA, a = i - b * AA;
  long off = (long)b * TT * AA + (long)t * AA + a;
  float v = (tf[t] > 0) ? aseq[off] : rec[off];
  inp[i] = (bf16)v;
}

// ---------------- WMMA GEMM: C[M,N] = X[M,K](lda) * W[N,K]^T (+bias)(+relu) ----------------
// One wave computes a 32x64 tile: 2 A fragments x 4 B fragments -> 8 WMMAs per 32-K slice,
// software-pipelined one K-slice ahead so loads overlap the XDL work.
// USE_LDS=1 (only when all 8 waves of a block share one 32-row A tile, nTiles%8==0):
//   A tile staged once via GLOBAL_LOAD_ASYNC_TO_LDS (ASYNCcnt) + s_wait_asynccnt + barrier;
//   A fragments then come from LDS as ds_load_b128 (template keeps the pointer chain purely
//   LDS-derived so infer-address-spaces emits DS ops, not flat). +16B row pad -> no bank
//   conflicts. B streams from L2.
// Requires: M%32==0, N%64==0, K%64==0. Grid = (M/32)*(N/64)/8 blocks of 256 threads.

__device__ __forceinline__ void load_a(const bf16* xr0, const bf16* xr1, int k, int kh,
                                       BFr& a0, BFr& a1) {
  a0.u[0] = *reinterpret_cast<const uint4*>(xr0 + k + kh * 8);
  a0.u[1] = *reinterpret_cast<const uint4*>(xr0 + k + kh * 8 + 16);
  a1.u[0] = *reinterpret_cast<const uint4*>(xr1 + k + kh * 8);
  a1.u[1] = *reinterpret_cast<const uint4*>(xr1 + k + kh * 8 + 16);
}

__device__ __forceinline__ void load_b(const bf16* w0, long wj, int k, int kh, BFr b[4]) {
  const bf16* wp = w0 + k + kh * 8;
#pragma unroll
  for (int j = 0; j < 4; j++) {
    b[j].u[0] = *reinterpret_cast<const uint4*>(wp + j * wj);
    b[j].u[1] = *reinterpret_cast<const uint4*>(wp + j * wj + 16);
  }
}

__device__ __forceinline__ void mma_slice(const BFr& a0, const BFr& a1,
                                          const BFr b[4], v8f acc[2][4]) {
#pragma unroll
  for (int j = 0; j < 4; j++) {
    acc[0][j] = __builtin_amdgcn_wmma_f32_16x16x32_bf16(
        false, a0.v, false, b[j].v, (short)0, acc[0][j], false, false);
    acc[1][j] = __builtin_amdgcn_wmma_f32_16x16x32_bf16(
        false, a1.v, false, b[j].v, (short)0, acc[1][j], false, false);
  }
}

extern __shared__ __align__(16) char lds_raw[];

template <int USE_LDS>
__global__ __launch_bounds__(256) void k_gemm_wmma(
    const bf16* __restrict__ X, int lda,
    const bf16* __restrict__ W, int K, int N, int M,
    const float* __restrict__ bias, int relu,
    float* __restrict__ outF, long ldF,
    bf16* __restrict__ outB, int ldB) {
  int wave = blockIdx.x * 8 + (threadIdx.x >> 5);
  int lane = threadIdx.x & 31;
  int nTiles = N >> 6;
  int mT = wave / nTiles;                  // 32-row tile index (block-uniform if USE_LDS)
  int nT = wave - mT * nTiles;
  int lm = lane & 15;
  int kh = lane >> 4;                      // which K-half this lane covers

  bf16* ldsA = (bf16*)lds_raw;             // [32][K+8] bf16 when USE_LDS
  const int LDP = K + 8;                   // row pitch: +16B pad -> bank-conflict-free

  if (USE_LDS) {
    // cooperative async stage of the block's 32xK A tile into LDS
    const bf16* src = X + (long)(mT * 32) * lda;
    int cpr = K >> 3;                      // 16B chunks per row
    int chunks = 32 * cpr;
    for (int c = threadIdx.x; c < chunks; c += 256) {
      int r = c / cpr, kc = c - r * cpr;
      const bf16* gp = src + (long)r * lda + (kc << 3);
      bf16* lp = ldsA + r * LDP + (kc << 3);
#ifdef HAVE_ASYNC_COPY
      __builtin_amdgcn_global_load_async_to_lds_b128((gv4i*)gp, (lv4i*)lp, 0, 0);
#else
      *reinterpret_cast<uint4*>(lp) = *reinterpret_cast<const uint4*>(gp);
#endif
    }
#ifdef HAVE_ASYNC_COPY
#  ifdef HAVE_WAIT_ASYNC
    __builtin_amdgcn_s_wait_asynccnt(0);
#  else
    asm volatile("s_wait_asynccnt 0x0" ::: "memory");
#  endif
#endif
    __syncthreads();
  }

  if (mT * 32 >= M) return;                // wave-uniform; never taken for exact grids

  // A-row pointers: purely LDS-derived (USE_LDS) or purely global -> no addrspace merge
  const bf16* xr0 = USE_LDS ? (ldsA + lm * LDP)
                            : (X + (long)(mT * 32 + lm) * lda);
  const bf16* xr1 = USE_LDS ? (ldsA + (lm + 16) * LDP)
                            : (X + (long)(mT * 32 + lm + 16) * lda);
  const bf16* w0 = W + (long)(nT * 64 + lm) * K;   // B: W rows, contiguous K
  const long  wj = (long)16 * K;                   // +16 output columns

  v8f acc[2][4] = {{{}, {}, {}, {}}, {{}, {}, {}, {}}};
  BFr aA0, aA1, bA[4], aB0, aB1, bB[4];

  load_a(xr0, xr1, 0, kh, aA0, aA1);
  load_b(w0, wj, 0, kh, bA);
  for (int k = 0; k < K; k += 64) {
    load_a(xr0, xr1, k + 32, kh, aB0, aB1);          // prefetch slice k+32
    load_b(w0, wj, k + 32, kh, bB);
    mma_slice(aA0, aA1, bA, acc);                    // compute slice k
    if (k + 64 < K) {
      load_a(xr0, xr1, k + 64, kh, aA0, aA1);        // prefetch slice k+64
      load_b(w0, wj, k + 64, kh, bA);
    }
    mma_slice(aB0, aB1, bB, acc);                    // compute slice k+32
  }

  // Epilogue. D layout: lane -> n = 16*j + lm ; vgpr v -> m = 32*mT + 16*mi + v + 8*kh
#pragma unroll
  for (int mi = 0; mi < 2; mi++) {
#pragma unroll
    for (int j = 0; j < 4; j++) {
      int n = nT * 64 + j * 16 + lm;
      float bv = bias ? bias[n] : 0.f;
#pragma unroll
      for (int v = 0; v < 8; v++) {
        int m = mT * 32 + mi * 16 + v + 8 * kh;
        float val = acc[mi][j][v] + bv;
        if (relu) val = fmaxf(val, 0.f);
        if (outF) outF[(long)m * ldF + n] = val;
        if (outB) outB[(long)m * ldB + n] = (bf16)val;
      }
    }
  }
}

// ---------------- host orchestration ----------------

extern "C" void kernel_launch(void* const* d_in, const int* in_sizes, int n_in,
                              void* d_out, int out_size, void* d_ws, size_t ws_size,
                              hipStream_t stream) {
  (void)in_sizes; (void)n_in; (void)out_size; (void)ws_size;
  const float* state  = (const float*)d_in[0];
  const float* aseq   = (const float*)d_in[1];
  const float* eps    = (const float*)d_in[2];
  const float* se_W   = (const float*)d_in[3];
  const float* se_b   = (const float*)d_in[4];
  const float* e0_Wih = (const float*)d_in[5];
  const float* e0_Whh = (const float*)d_in[6];
  const float* e0_bih = (const float*)d_in[7];
  const float* e0_bhh = (const float*)d_in[8];
  const float* e1_Wih = (const float*)d_in[9];
  const float* e1_Whh = (const float*)d_in[10];
  const float* e1_bih = (const float*)d_in[11];
  const float* e1_bhh = (const float*)d_in[12];
  const float* ce_W   = (const float*)d_in[13];
  const float* ce_b   = (const float*)d_in[14];
  const float* mu_W   = (const float*)d_in[15];
  const float* mu_b   = (const float*)d_in[16];
  const float* lv_W   = (const float*)d_in[17];
  const float* lv_b   = (const float*)d_in[18];
  const float* di_W   = (const float*)d_in[19];
  const float* di_b   = (const float*)d_in[20];
  const float* dc_Wih = (const float*)d_in[21];
  const float* dc_Whh = (const float*)d_in[22];
  const float* dc_bih = (const float*)d_in[23];
  const float* dc_bhh = (const float*)d_in[24];
  const float* a1_W   = (const float*)d_in[25];
  const float* a1_b   = (const float*)d_in[26];
  const float* a2_W   = (const float*)d_in[27];
  const float* a2_b   = (const float*)d_in[28];
  const float* initA  = (const float*)d_in[29];
  const int*   tf     = (const int*)d_in[30];

  float* out    = (float*)d_out;
  float* outRec = out;                                  // [B,T,A]
  float* outMu  = out + (size_t)BB * TT * AA;           // [B,L]
  float* outLv  = outMu + (size_t)BB * LL;              // [B,L]

  char* wsB = (char*)d_ws;
  size_t cur = 0;
  auto alloc = [&](size_t bytes) -> char* {
    char* p = wsB + cur;
    cur += (bytes + 255) & ~(size_t)255;
    return p;
  };
  bf16* w_se  = (bf16*)alloc(2ull * HH * SDD);
  bf16* w_e0i = (bf16*)alloc(2ull * H3 * AA);
  bf16* w_e0h = (bf16*)alloc(2ull * H3 * HH);
  bf16* w_e1i = (bf16*)alloc(2ull * H3 * HH);
  bf16* w_e1h = (bf16*)alloc(2ull * H3 * HH);
  bf16* w_ce  = (bf16*)alloc(2ull * HH * 2 * HH);
  bf16* w_mu  = (bf16*)alloc(2ull * LL * HH);
  bf16* w_lv  = (bf16*)alloc(2ull * LL * HH);
  bf16* w_di  = (bf16*)alloc(2ull * HH * (LL + SDD));
  bf16* w_dci = (bf16*)alloc(2ull * H3 * AA);
  bf16* w_dch = (bf16*)alloc(2ull * H3 * HH);
  bf16* w_a1  = (bf16*)alloc(2ull * HH * HH);
  bf16* w_a2  = (bf16*)alloc(2ull * AA * HH);
  bf16* st_bf = (bf16*)alloc(2ull * BB * SDD);
  bf16* as_bf = (bf16*)alloc(2ull * BB * TT * AA);
  bf16* h0bf  = (bf16*)alloc(2ull * BB * HH);
  bf16* h1bf  = (bf16*)alloc(2ull * BB * HH);
  bf16* cat2h = (bf16*)alloc(2ull * BB * 2 * HH);
  bf16* hid   = (bf16*)alloc(2ull * BB * HH);
  bf16* zcat  = (bf16*)alloc(2ull * BB * (LL + SDD));
  bf16* dhbf  = (bf16*)alloc(2ull * BB * HH);
  bf16* inp   = (bf16*)alloc(2ull * BB * AA);
  bf16* ahh   = (bf16*)alloc(2ull * BB * HH);
  float* gi   = (float*)alloc(4ull * BB * H3);
  float* gh   = (float*)alloc(4ull * BB * H3);
  float* h0f  = (float*)alloc(4ull * BB * HH);
  float* h1f  = (float*)alloc(4ull * BB * HH);
  float* dhf  = (float*)alloc(4ull * BB * HH);

  auto cvt = [&](const float* s, bf16* d, long n) {
    k_cvt<<<dim3((unsigned)((n + 255) / 256)), dim3(256), 0, stream>>>(s, d, (int)n);
  };
  auto zero = [&](void* p, long bytes) {
    k_zero<<<dim3((unsigned)((bytes / 4 + 255) / 256)), dim3(256), 0, stream>>>(
        (unsigned int*)p, (int)(bytes / 4));
  };
  // waves = (M/32)*(N/64); blocks = waves/8 = N/32 for M=512 (exact for all N used).
  // LDS staging only when all 8 waves of a block share one A tile (nTiles%8==0).
  auto gemm = [&](const bf16* X, int lda, const bf16* W, int N, int K,
                  const float* bias, int relu, float* oF, long ldF, bf16* oB, int ldB) {
    if ((((N >> 6) & 7) == 0)) {
      size_t shmem = (size_t)32 * (K + 8) * 2;
      k_gemm_wmma<1><<<dim3((unsigned)(N / 32)), dim3(256), shmem, stream>>>(
          X, lda, W, K, N, BB, bias, relu, oF, ldF, oB, ldB);
    } else {
      k_gemm_wmma<0><<<dim3((unsigned)(N / 32)), dim3(256), 0, stream>>>(
          X, lda, W, K, N, BB, bias, relu, oF, ldF, oB, ldB);
    }
  };

  // one-time bf16 conversion (recomputed every call; no cached state)
  cvt(se_W, w_se, HH * SDD);
  cvt(e0_Wih, w_e0i, (long)H3 * AA);
  cvt(e0_Whh, w_e0h, (long)H3 * HH);
  cvt(e1_Wih, w_e1i, (long)H3 * HH);
  cvt(e1_Whh, w_e1h, (long)H3 * HH);
  cvt(ce_W, w_ce, (long)HH * 2 * HH);
  cvt(mu_W, w_mu, (long)LL * HH);
  cvt(lv_W, w_lv, (long)LL * HH);
  cvt(di_W, w_di, (long)HH * (LL + SDD));
  cvt(dc_Wih, w_dci, (long)H3 * AA);
  cvt(dc_Whh, w_dch, (long)H3 * HH);
  cvt(a1_W, w_a1, (long)HH * HH);
  cvt(a2_W, w_a2, (long)AA * HH);
  cvt(state, st_bf, (long)BB * SDD);
  cvt(aseq, as_bf, (long)BB * TT * AA);
  zero(h0f, 4ull * BB * HH);
  zero(h1f, 4ull * BB * HH);
  zero(h0bf, 2ull * BB * HH);
  zero(h1bf, 2ull * BB * HH);

  const dim3 gBH((BB * HH) / 256), gBA((BB * AA) / 256), b256(256);

  // ---- stacked GRU encoder (enc1 step t consumes enc0's h after step t) ----
  for (int t = 0; t < TT; t++) {
    gemm(as_bf + (long)t * AA, TT * AA, w_e0i, H3, AA, nullptr, 0, gi, H3, nullptr, 0);
    gemm(h0bf, HH, w_e0h, H3, HH, nullptr, 0, gh, H3, nullptr, 0);
    k_gru_gate<<<gBH, b256, 0, stream>>>(gi, gh, e0_bih, e0_bhh, h0f, h0bf);
    gemm(h0bf, HH, w_e1i, H3, HH, nullptr, 0, gi, H3, nullptr, 0);
    gemm(h1bf, HH, w_e1h, H3, HH, nullptr, 0, gh, H3, nullptr, 0);
    k_gru_gate<<<gBH, b256, 0, stream>>>(gi, gh, e1_bih, e1_bhh, h1f, h1bf);
  }

  // ---- VAE head ----
  gemm(st_bf, SDD, w_se, HH, SDD, se_b, 1, nullptr, 0, cat2h, 2 * HH);        // s_enc
  k_copy_bf<<<gBH, b256, 0, stream>>>(h1bf, HH, cat2h + HH, 2 * HH, BB, HH);  // [s_enc,hT]
  gemm(cat2h, 2 * HH, w_ce, HH, 2 * HH, ce_b, 1, nullptr, 0, hid, HH);
  gemm(hid, HH, w_mu, LL, HH, mu_b, 0, outMu, LL, nullptr, 0);
  gemm(hid, HH, w_lv, LL, HH, lv_b, 0, outLv, LL, nullptr, 0);
  k_reparam<<<dim3((BB * (LL + SDD)) / 256), b256, 0, stream>>>(outMu, outLv, eps, state, zcat);

  // ---- decoder ----
  gemm(zcat, LL + SDD, w_di, HH, LL + SDD, di_b, 1, dhf, HH, dhbf, HH);       // dh0
  k_initinp<<<gBA, b256, 0, stream>>>(initA, inp);
  for (int t = 0; t < TT; t++) {
    gemm(inp, AA, w_dci, H3, AA, nullptr, 0, gi, H3, nullptr, 0);
    gemm(dhbf, HH, w_dch, H3, HH, nullptr, 0, gh, H3, nullptr, 0);
    k_gru_gate<<<gBH, b256, 0, stream>>>(gi, gh, dc_bih, dc_bhh, dhf, dhbf);
    gemm(dhbf, HH, w_a1, HH, HH, a1_b, 1, nullptr, 0, ahh, HH);
    gemm(ahh, HH, w_a2, AA, HH, a2_b, 0, outRec + (long)t * AA, (long)TT * AA, nullptr, 0);
    if (t + 1 < TT)
      k_nextinp<<<gBA, b256, 0, stream>>>(outRec, aseq, tf, t, inp);
  }
}
